// SelfAttention2d_57690000720167
// MI455X (gfx1250) — matrix operand order repaired and checked by reading, not compiled, not run on hardware
//
#include <hip/hip_runtime.h>

typedef unsigned short u16;
typedef unsigned int   u32;
typedef unsigned long long u64;

typedef __attribute__((ext_vector_type(16))) __bf16 v16bf;
typedef __attribute__((ext_vector_type(8)))  float  v8f;
typedef __attribute__((ext_vector_type(4)))  u32    v4u;
typedef __attribute__((ext_vector_type(8)))  u32    v8u;

#define B_      4
#define C_      256
#define N_      4096
#define G_      8
#define HEADS_  4
#define D_      64
#define THREEC_ 768
#define SCALE_Q 0.125f   // 64^-0.5

__device__ __forceinline__ u16 f2bf(float f) {
  u32 u = __float_as_uint(f);
  u32 r = u + 0x7FFFu + ((u >> 16) & 1u);   // round-to-nearest-even
  return (u16)(r >> 16);
}

union FragU { v16bf v; uint4 q[2]; };

// A-fragment (16x32 bf16): lane holds row M=lane%16; K halves by lane half
// (0..7/16..23 vs 8..15/24..31) -> two contiguous 16B LDS loads.
__device__ __forceinline__ v16bf load_frag_A(const u16* t, int ld, int m_base, int k_base) {
  int lane = threadIdx.x & 31;
  int row  = m_base + (lane & 15);
  int kh   = (lane >> 4) << 3;
  FragU f;
  f.q[0] = *(const uint4*)(t + row * ld + k_base + kh);
  f.q[1] = *(const uint4*)(t + row * ld + k_base + 16 + kh);
  return f.v;
}

// B-fragment (32x16 bf16) from [n][k]-transposed LDS tile: lane holds col
// n=lane%16, K halves by lane half (0..7/16..23 vs 8..15/24..31) -> 2x b128.
__device__ __forceinline__ v16bf load_frag_B(const u16* t, int ld, int n_base, int k_base) {
  int lane = threadIdx.x & 31;
  int row  = n_base + (lane & 15);
  int kh   = (lane >> 4) << 3;
  FragU f;
  f.q[0] = *(const uint4*)(t + row * ld + k_base + kh);
  f.q[1] = *(const uint4*)(t + row * ld + k_base + 16 + kh);
  return f.v;
}

__device__ __forceinline__ v8f wmma_bf16(v16bf a, v16bf b, v8f c) {
  return __builtin_amdgcn_wmma_f32_16x16x32_bf16(false, a, false, b, (short)0, c, false, false);
}

__device__ __forceinline__ void unpack8(uint4 v, u16* s) {
  s[0] = (u16)(v.x); s[1] = (u16)(v.x >> 16);
  s[2] = (u16)(v.y); s[3] = (u16)(v.y >> 16);
  s[4] = (u16)(v.z); s[5] = (u16)(v.z >> 16);
  s[6] = (u16)(v.w); s[7] = (u16)(v.w >> 16);
}

// ---- DPP16 butterfly reductions over a 16-lane row, fused into max/add ----
// XOR masks {1,2,7,15}: quad_perm[1,0,3,2], quad_perm[2,3,0,1],
// row_half_mirror, row_mirror span all 16 lanes of a row. 1 VALU op per step.
__device__ __forceinline__ float red_max16(float v) {
  asm("v_max_num_f32_dpp %0, %0, %0 quad_perm:[1,0,3,2] row_mask:0xf bank_mask:0xf bound_ctrl:1" : "+v"(v));
  asm("v_max_num_f32_dpp %0, %0, %0 quad_perm:[2,3,0,1] row_mask:0xf bank_mask:0xf bound_ctrl:1" : "+v"(v));
  asm("v_max_num_f32_dpp %0, %0, %0 row_half_mirror row_mask:0xf bank_mask:0xf bound_ctrl:1" : "+v"(v));
  asm("v_max_num_f32_dpp %0, %0, %0 row_mirror row_mask:0xf bank_mask:0xf bound_ctrl:1" : "+v"(v));
  return v;
}
__device__ __forceinline__ float red_sum16(float v) {
  asm("v_add_f32_dpp %0, %0, %0 quad_perm:[1,0,3,2] row_mask:0xf bank_mask:0xf bound_ctrl:1" : "+v"(v));
  asm("v_add_f32_dpp %0, %0, %0 quad_perm:[2,3,0,1] row_mask:0xf bank_mask:0xf bound_ctrl:1" : "+v"(v));
  asm("v_add_f32_dpp %0, %0, %0 row_half_mirror row_mask:0xf bank_mask:0xf bound_ctrl:1" : "+v"(v));
  asm("v_add_f32_dpp %0, %0, %0 row_mirror row_mask:0xf bank_mask:0xf bound_ctrl:1" : "+v"(v));
  return v;
}

// ---- Tensor Data Mover: 2D tile load Global -> LDS with LDS row padding ----
// D# per CDNA5 ISA 8.3/8.4. data_size=2B. tile_x contiguous elements per row,
// tile_y rows, global row stride stride_x elements. pad codes: interval 3 =
// every 16 DWORDs (64B = 32 elems), amount 3 = 4 DWORDs (8 elems) -> ld 40.
__device__ __forceinline__ void tdm_load_2d(u32 lds_off, const void* gaddr,
                                            u32 tile_x, u32 tile_y,
                                            u32 tensor_x, u32 tensor_y,
                                            u32 stride_x,
                                            u32 pad_int_code, u32 pad_amt_code) {
  u64 ga = (u64)gaddr;
  v4u g0;
  g0[0] = 1u;                                   // count=1, user descriptor
  g0[1] = lds_off;                              // lds_addr (bytes)
  g0[2] = (u32)ga;                              // global_addr[31:0]
  g0[3] = (u32)(ga >> 32) | (2u << 30);         // global_addr[56:32], type=2
  v8u g1;
  g1[0] = (1u << 16) | (1u << 20) | (pad_int_code << 22) | (pad_amt_code << 25);
  g1[1] = (tensor_x & 0xFFFFu) << 16;           // tensor_dim0[15:0]
  g1[2] = (tensor_x >> 16) | ((tensor_y & 0xFFFFu) << 16);
  g1[3] = (tensor_y >> 16) | (tile_x << 16);    // tile_dim0
  g1[4] = tile_y;                               // tile_dim1 (tile_dim2 = 0)
  g1[5] = stride_x;                             // tensor_dim0_stride[31:0]
  g1[6] = 0u;
  g1[7] = 0u;
  asm volatile("tensor_load_to_lds %0, %1" :: "s"(g0), "s"(g1) : "memory");
}
__device__ __forceinline__ void tdm_wait0() {
#if __has_builtin(__builtin_amdgcn_s_wait_tensorcnt)
  __builtin_amdgcn_s_wait_tensorcnt(0);
#else
  asm volatile("s_wait_tensorcnt 0x0" ::: "memory");
#endif
}

// ---------------- GroupNorm ----------------

__global__ __launch_bounds__(256)
void gn_stats_kernel(const float* __restrict__ x, float* __restrict__ stats) {
  __shared__ float rs[256], rq[256];
  const int bg = blockIdx.x;                          // b*8 + g
  const float4* p = (const float4*)(x + (size_t)bg * 32 * N_);
  float s = 0.f, sq = 0.f;
  for (int i = threadIdx.x; i < (32 * N_) / 4; i += 256) {
    float4 v = p[i];
    s  += v.x + v.y + v.z + v.w;
    sq += v.x * v.x + v.y * v.y + v.z * v.z + v.w * v.w;
  }
  rs[threadIdx.x] = s; rq[threadIdx.x] = sq;
  __syncthreads();
  for (int o = 128; o > 0; o >>= 1) {
    if ((int)threadIdx.x < o) {
      rs[threadIdx.x] += rs[threadIdx.x + o];
      rq[threadIdx.x] += rq[threadIdx.x + o];
    }
    __syncthreads();
  }
  if (threadIdx.x == 0) {
    const float cnt = (float)(32 * N_);
    float mean = rs[0] / cnt;
    float var  = rq[0] / cnt - mean * mean;
    stats[2 * bg]     = mean;
    stats[2 * bg + 1] = rsqrtf(var + 1e-5f);
  }
}

__global__ __launch_bounds__(256)
void gn_apply_kernel(const float* __restrict__ x, const float* __restrict__ gamma,
                     const float* __restrict__ beta, const float* __restrict__ stats,
                     u16* __restrict__ xn) {
  const int bc = blockIdx.x;                 // b*256 + c
  const int b = bc >> 8, c = bc & 255;
  const int g = c >> 5;
  const float mean = stats[2 * (b * G_ + g)];
  const float rstd = stats[2 * (b * G_ + g) + 1];
  const float ga = gamma[c] * rstd;
  const float be = beta[c] - mean * ga;
  const float4* px = (const float4*)(x + (size_t)bc * N_);
  u16* po = xn + (size_t)bc * N_;
  for (int i = threadIdx.x; i < N_ / 4; i += 256) {
    float4 v = px[i];
    u32 lo = (u32)f2bf(v.x * ga + be) | ((u32)f2bf(v.y * ga + be) << 16);
    u32 hi = (u32)f2bf(v.z * ga + be) | ((u32)f2bf(v.w * ga + be) << 16);
    uint2 t; t.x = lo; t.y = hi;
    *(uint2*)(po + i * 4) = t;
  }
}

__global__ void f32_to_bf16_kernel(const float* __restrict__ in, u16* __restrict__ out, int n) {
  int i = blockIdx.x * 256 + threadIdx.x;
  if (i < n) out[i] = f2bf(in[i]);
}

// ---------------- GEMM: out[o][n] = sum_c A[o][c] * B[c][n] + bias[o] ----------------
// Block tile 64(M) x 128(N), K staged 64-deep per barrier. 8 waves (2x4),
// 32x32 per wave -> 8 WMMA per stage.
// MODE 0: QKV -> bf16 out, q-rows (o<256) pre-scaled by SCALE_Q
// MODE 1: proj -> f32 out + residual

template <int MODE>
__global__ __launch_bounds__(256)
void gemm_bf16_kernel(const u16* __restrict__ A, const u16* __restrict__ Bm,
                      const float* __restrict__ bias,
                      u16* __restrict__ out_bf, const float* __restrict__ resid,
                      float* __restrict__ out_f32, int Ktot) {
  __shared__ __align__(16) u16 At[64 * 72];     // [m][k], ld=72
  __shared__ __align__(16) u16 Bt[128 * 72];    // [n][k] transposed, ld=72
  const int bn = blockIdx.x * 128;
  const int bm = blockIdx.y * 64;
  const int batch = blockIdx.z;
  const u16* Bb = Bm + (size_t)batch * Ktot * N_;
  const int wave = threadIdx.x >> 5;
  const int wm = wave >> 2, wn = wave & 3;
  v8f c00 = {}, c01 = {}, c10 = {}, c11 = {};

  for (int kk = 0; kk < Ktot; kk += 64) {
    #pragma unroll
    for (int p = 0; p < 2; ++p) {               // A: 64x64 = 512 uint4
      int l = threadIdx.x + p * 256;
      int r = l >> 3, c8 = (l & 7) << 3;
      uint4 va = *(const uint4*)(A + (size_t)(bm + r) * Ktot + kk + c8);
      *(uint4*)(At + r * 72 + c8) = va;
    }
    #pragma unroll
    for (int p = 0; p < 4; ++p) {               // B: 64x128 = 1024 uint4, transpose
      int q = threadIdx.x + p * 256;
      int k = q >> 4, jg = q & 15;
      uint4 vb = *(const uint4*)(Bb + (size_t)(kk + k) * N_ + bn + jg * 8);
      u16 s[8]; unpack8(vb, s);
      #pragma unroll
      for (int u = 0; u < 8; ++u) Bt[(jg * 8 + u) * 72 + k] = s[u];
    }
    if (kk + 64 < Ktot) {                       // prefetch next B stage
      int q = threadIdx.x;
      int k = q >> 4, jg = q & 15;
      __builtin_prefetch(Bb + (size_t)(kk + 64 + k) * N_ + bn + jg * 8, 0, 1);
    }
    __syncthreads();
    #pragma unroll
    for (int kc = 0; kc < 64; kc += 32) {
      v16bf a0 = load_frag_A(At, 72, wm * 32, kc);
      v16bf a1 = load_frag_A(At, 72, wm * 32 + 16, kc);
      v16bf b0 = load_frag_B(Bt, 72, wn * 32, kc);
      v16bf b1 = load_frag_B(Bt, 72, wn * 32 + 16, kc);
      c00 = wmma_bf16(a0, b0, c00);
      c01 = wmma_bf16(a0, b1, c01);
      c10 = wmma_bf16(a1, b0, c10);
      c11 = wmma_bf16(a1, b1, c11);
    }
    __syncthreads();
  }

  const int lane = threadIdx.x & 31;
  const int coll = lane & 15, half = lane >> 4;
  v8f cc[2][2] = {{c00, c01}, {c10, c11}};
  #pragma unroll
  for (int i = 0; i < 2; ++i)
    #pragma unroll
    for (int j = 0; j < 2; ++j) {
      int n = bn + wn * 32 + j * 16 + coll;
      #pragma unroll
      for (int r = 0; r < 8; ++r) {
        int o = bm + wm * 32 + i * 16 + half * 8 + r;
        float v = cc[i][j][r] + bias[o];
        if (MODE == 0) {
          if (o < C_) v *= SCALE_Q;
          out_bf[(size_t)(batch * THREEC_ + o) * N_ + n] = f2bf(v);
        } else {
          size_t idx = (size_t)(batch * C_ + o) * N_ + n;
          out_f32[idx] = v + resid[idx];
        }
      }
    }
}

// ---------------- Flash attention ----------------
// qkv layout: [b][o][n], o = {0:q,1:k,2:v}*256 + head*64 + d. Block = 128 queries,
// 8 waves of 16 queries each; 32-key tiles streamed through LDS. V tile is
// staged by the Tensor Data Mover (async DMA + LDS pad), K by manual transpose.

__global__ __launch_bounds__(256)
void attn_kernel(const u16* __restrict__ qkv, u16* __restrict__ attn_out) {
  __shared__ __align__(16) u16 QT[128 * 72];    // [query][d], ld=72
  __shared__ __align__(16) u16 KT[32 * 72];     // [key][d],   ld=72
  __shared__ __align__(16) u16 VT[64 * 40];     // [d][key],   ld=40 (TDM pad)
  __shared__ __align__(16) u16 PT[8 * 16 * 40]; // per-wave P tile [row][key]
  const int nblk = blockIdx.x * 128;
  const int head = blockIdx.y;
  const int b    = blockIdx.z;
  const u16* q_ptr = qkv + ((size_t)b * THREEC_ + head * D_) * N_;
  const u16* k_ptr = qkv + ((size_t)b * THREEC_ + C_ + head * D_) * N_;
  const u16* v_ptr = qkv + ((size_t)b * THREEC_ + 2 * C_ + head * D_) * N_;

  // Stage Q transposed: global [d][n] -> QT[n-local][d]
  #pragma unroll
  for (int i = 0; i < 4; ++i) {
    int l = threadIdx.x + i * 256;              // 0..1023
    int d = l >> 4, grp = l & 15;
    uint4 v = *(const uint4*)(q_ptr + (size_t)d * N_ + nblk + grp * 8);
    u16 s[8]; unpack8(v, s);
    #pragma unroll
    for (int u = 0; u < 8; ++u) QT[(grp * 8 + u) * 72 + d] = s[u];
  }
  __syncthreads();

  const int wave = threadIdx.x >> 5;
  const int lane = threadIdx.x & 31;
  const int coll = lane & 15, half = lane >> 4;
  v16bf aq0 = load_frag_A(QT, 72, wave * 16, 0);    // d = 0..31
  v16bf aq1 = load_frag_A(QT, 72, wave * 16, 32);   // d = 32..63
  u16* Pw = PT + wave * 16 * 40;
  const u32 vt_lds = (u32)(size_t)(void*)VT;

  float m_run[8], l_run[8];
  v8f o0 = {}, o1 = {}, o2 = {}, o3 = {};
  #pragma unroll
  for (int r = 0; r < 8; ++r) { m_run[r] = -1e30f; l_run[r] = 0.f; }

  for (int kt = 0; kt < N_; kt += 32) {
    __syncthreads();   // previous tile fully consumed
    if (wave == 0) {   // async DMA: V tile [64 d][32 keys] -> VT (ld 40)
      tdm_load_2d(vt_lds, v_ptr + kt, /*tile_x=*/32, /*tile_y=*/64,
                  /*tensor_x=*/N_, /*tensor_y=*/64, /*stride_x=*/N_,
                  /*pad_int=*/3, /*pad_amt=*/3);
    }
    {
      int l = threadIdx.x;
      int d = l >> 2, grp = l & 3;
      uint4 kv = *(const uint4*)(k_ptr + (size_t)d * N_ + kt + grp * 8);
      u16 s[8]; unpack8(kv, s);
      #pragma unroll
      for (int u = 0; u < 8; ++u) KT[(grp * 8 + u) * 72 + d] = s[u];
      int ktn = (kt + 32 < N_) ? (kt + 32) : kt;  // prefetch next tile
      __builtin_prefetch(k_ptr + (size_t)d * N_ + ktn + grp * 8, 0, 1);
      __builtin_prefetch(v_ptr + (size_t)d * N_ + ktn + grp * 8, 0, 1);
    }
    if (wave == 0) tdm_wait0();
    __syncthreads();

    // S = Q * K^T (16 queries x 32 keys), fp32 accumulate
    v8f s0 = {}, s1 = {};
    s0 = wmma_bf16(aq0, load_frag_B(KT, 72, 0, 0),  s0);
    s0 = wmma_bf16(aq1, load_frag_B(KT, 72, 0, 32), s0);
    s1 = wmma_bf16(aq0, load_frag_B(KT, 72, 16, 0),  s1);
    s1 = wmma_bf16(aq1, load_frag_B(KT, 72, 16, 32), s1);

    // Online softmax: row r of lane maps to query (r + 8*half). DPP16 reductions.
    float mnew[8], alpha[8];
    #pragma unroll
    for (int r = 0; r < 8; ++r) {
      float v = red_max16(fmaxf(s0[r], s1[r]));
      mnew[r]  = fmaxf(m_run[r], v);
      alpha[r] = __expf(m_run[r] - mnew[r]);
    }
    float p0[8], p1[8];
    #pragma unroll
    for (int r = 0; r < 8; ++r) {
      p0[r] = __expf(s0[r] - mnew[r]);
      p1[r] = __expf(s1[r] - mnew[r]);
      float sum = red_sum16(p0[r] + p1[r]);
      l_run[r] = l_run[r] * alpha[r] + sum;
      m_run[r] = mnew[r];
      o0[r] *= alpha[r]; o1[r] *= alpha[r]; o2[r] *= alpha[r]; o3[r] *= alpha[r];
    }

    // P -> wave-private LDS tile [row][key], then reload as A-fragment
    #pragma unroll
    for (int r = 0; r < 8; ++r) {
      Pw[(half * 8 + r) * 40 + coll]      = f2bf(p0[r]);
      Pw[(half * 8 + r) * 40 + 16 + coll] = f2bf(p1[r]);
    }
    v16bf pa = load_frag_A(Pw, 40, 0, 0);
    o0 = wmma_bf16(pa, load_frag_B(VT, 40, 0, 0),  o0);
    o1 = wmma_bf16(pa, load_frag_B(VT, 40, 16, 0), o1);
    o2 = wmma_bf16(pa, load_frag_B(VT, 40, 32, 0), o2);
    o3 = wmma_bf16(pa, load_frag_B(VT, 40, 48, 0), o3);
  }

  // Epilogue: out[b][head*64 + dcol][n], 8 consecutive n per lane -> b128 stores
  float inv_l[8];
  #pragma unroll
  for (int r = 0; r < 8; ++r) inv_l[r] = 1.f / l_run[r];
  v8f oo[4] = {o0, o1, o2, o3};
  #pragma unroll
  for (int c = 0; c < 4; ++c) {
    u32 pk[4];
    #pragma unroll
    for (int r = 0; r < 4; ++r) {
      u32 lo = f2bf(oo[c][2 * r]     * inv_l[2 * r]);
      u32 hi = f2bf(oo[c][2 * r + 1] * inv_l[2 * r + 1]);
      pk[r] = lo | (hi << 16);
    }
    uint4 st; st.x = pk[0]; st.y = pk[1]; st.z = pk[2]; st.w = pk[3];
    size_t off = (size_t)(b * C_ + head * D_ + c * 16 + coll) * N_ + nblk + wave * 16 + half * 8;
    *(uint4*)(attn_out + off) = st;
  }
}

// ---------------- host launch ----------------

extern "C" void kernel_launch(void* const* d_in, const int* in_sizes, int n_in,
                              void* d_out, int out_size, void* d_ws, size_t ws_size,
                              hipStream_t stream) {
  (void)in_sizes; (void)n_in; (void)out_size; (void)ws_size;
  const float* x      = (const float*)d_in[0];
  const float* gamma  = (const float*)d_in[1];
  const float* beta   = (const float*)d_in[2];
  const float* w_qkv  = (const float*)d_in[3];
  const float* b_qkv  = (const float*)d_in[4];
  const float* w_proj = (const float*)d_in[5];
  const float* b_proj = (const float*)d_in[6];
  float* out = (float*)d_out;

  char* ws = (char*)d_ws;
  float* stats  = (float*)(ws);                                  // 256 B
  u16* wqkv_bf  = (u16*)(ws + 256);                              // 768*256*2
  u16* wproj_bf = (u16*)(ws + 256 + 768 * 256 * 2);              // 256*256*2
  size_t off = 524544;                                           // 16B aligned
  u16* xn   = (u16*)(ws + off); off += (size_t)B_ * C_ * N_ * 2;
  u16* qkv  = (u16*)(ws + off); off += (size_t)B_ * THREEC_ * N_ * 2;
  u16* attn = (u16*)(ws + off);                                  // + B*C*N*2

  gn_stats_kernel<<<B_ * G_, 256, 0, stream>>>(x, stats);
  f32_to_bf16_kernel<<<(768 * 256 + 255) / 256, 256, 0, stream>>>(w_qkv, wqkv_bf, 768 * 256);
  f32_to_bf16_kernel<<<(256 * 256 + 255) / 256, 256, 0, stream>>>(w_proj, wproj_bf, 256 * 256);
  gn_apply_kernel<<<B_ * C_, 256, 0, stream>>>(x, gamma, beta, stats, xn);
  gemm_bf16_kernel<0><<<dim3(N_ / 128, THREEC_ / 64, B_), 256, 0, stream>>>(
      wqkv_bf, xn, b_qkv, qkv, nullptr, nullptr, C_);
  attn_kernel<<<dim3(N_ / 128, HEADS_, B_), 256, 0, stream>>>(qkv, attn);
  gemm_bf16_kernel<1><<<dim3(N_ / 128, C_ / 64, B_), 256, 0, stream>>>(
      wproj_bf, attn, b_proj, nullptr, x, out, C_);
}
